// CronRootAttentionV14_50929722196092
// MI455X (gfx1250) — matrix operand order repaired and checked by reading, not compiled, not run on hardware
//
#include <hip/hip_runtime.h>

// Problem constants (match reference)
#define BB   2
#define HH   16
#define SS   4096
#define DD   128
#define WW   64
#define NBLK 64
#define RR   64

// Padded LDS row strides (element units). All chosen so that:
//  - rows stagger across LDS banks (stride % 64 dwords != 0)
//  - every fragment b128 access stays 16-byte aligned
#define SQ_STR  136   // bf16, Q rows      (68 dwords)
#define SK_STR  136   // bf16, K rows      (68 dwords)
#define SV_STR  264   // bf16, Vt rows     (132 dwords)
#define SS_STR  260   // f32,  score rows  (260 dwords)
#define SP_STR  264   // bf16, P rows      (132 dwords)

typedef __attribute__((ext_vector_type(16))) __bf16 v16bf;
typedef __attribute__((ext_vector_type(8)))  float  v8f;

union FragU {
    uint4        q[2];
    unsigned int u[8];
    v16bf        v;
};

// Native bf16 converts (RNE). clang lowers these to the gfx1250 hardware
// cvt ops (packed v_cvt_pk_bf16_f32 where it can) instead of a 4-5 VALU
// software rounding sequence.
__device__ __forceinline__ unsigned short f2bf(float a) {
    __bf16 h = (__bf16)a;
    return __builtin_bit_cast(unsigned short, h);
}
__device__ __forceinline__ unsigned int pack2_bf16(float a, float b) {
    return (unsigned int)f2bf(a) | ((unsigned int)f2bf(b) << 16);
}

// LDS layout (dynamic shared):
//   sQ  : bf16 [64][136]   @ 0        17408 B
//   sK  : bf16 [256][136]  @ 17408    69632 B
//   sVt : bf16 [128][264]  @ 87040    67584 B   (V transposed: [d][key])
//   sS  : f32  [64][260]   @ 154624   66560 B
//   sP  : bf16 [64][264]   @ 221184   33792 B
#define OFF_Q  0
#define OFF_K  17408
#define OFF_V  87040
#define OFF_S  154624
#define OFF_P  221184
#define SMEM_BYTES 254976

__global__ __launch_bounds__(256)
void cron_root_attn_wmma(const float* __restrict__ q, const float* __restrict__ k,
                         const float* __restrict__ v, const float* __restrict__ rk,
                         const float* __restrict__ rv, float* __restrict__ out) {
    extern __shared__ char smem[];
    unsigned short* sQ  = (unsigned short*)(smem + OFF_Q);
    unsigned short* sK  = (unsigned short*)(smem + OFF_K);
    unsigned short* sVt = (unsigned short*)(smem + OFF_V);
    float*          sS  = (float*)(smem + OFF_S);
    unsigned short* sP  = (unsigned short*)(smem + OFF_P);

    const int tid = threadIdx.x;
    const int bid = blockIdx.x;
    const int g = bid % NBLK;
    const int h = (bid / NBLK) % HH;
    const int b = bid / (NBLK * HH);

    const size_t bh = (size_t)(b * HH + h);
    const float* qbh  = q  + bh * (size_t)SS * DD;
    const float* kbh  = k  + bh * (size_t)SS * DD;
    const float* vbh  = v  + bh * (size_t)SS * DD;
    const float* rkbh = rk + bh * (size_t)RR * DD;
    const float* rvbh = rv + bh * (size_t)RR * DD;
    float*       obh  = out + bh * (size_t)SS * DD;

    // ---------------- Stage Q (64x128) as bf16 ----------------
    for (int idx = tid; idx < 64 * 32; idx += 256) {
        int row = idx >> 5;
        int d   = (idx & 31) << 2;
        const float4 f = *(const float4*)(qbh + (size_t)(g * WW + row) * DD + d);
        unsigned int* dst = (unsigned int*)(sQ + row * SQ_STR + d);
        dst[0] = pack2_bf16(f.x, f.y);
        dst[1] = pack2_bf16(f.z, f.w);
    }

    // ---------------- Stage 256-key K set + V set (V transposed) ----------------
    // key rows: [0,64)=prev block, [64,128)=current block, [128,192)=strided, [192,256)=rel
    for (int idx = tid; idx < 256 * 32; idx += 256) {
        int row = idx >> 5;
        int d   = (idx & 31) << 2;
        float4 fk, fv;
        if (row < 64) {
            if (g == 0) { fk = make_float4(0.f,0.f,0.f,0.f); fv = fk; }
            else {
                size_t off = (size_t)((g - 1) * WW + row) * DD + d;
                fk = *(const float4*)(kbh + off);
                fv = *(const float4*)(vbh + off);
            }
        } else if (row < 128) {
            size_t off = (size_t)(g * WW + (row - 64)) * DD + d;
            fk = *(const float4*)(kbh + off);
            fv = *(const float4*)(vbh + off);
        } else if (row < 192) {
            size_t off = (size_t)((row - 128) * WW) * DD + d;   // block starts ::W
            fk = *(const float4*)(kbh + off);
            fv = *(const float4*)(vbh + off);
        } else {
            size_t off = (size_t)(row - 192) * DD + d;
            fk = *(const float4*)(rkbh + off);
            fv = *(const float4*)(rvbh + off);
        }
        unsigned int* dk = (unsigned int*)(sK + row * SK_STR + d);
        dk[0] = pack2_bf16(fk.x, fk.y);
        dk[1] = pack2_bf16(fk.z, fk.w);
        sVt[(d + 0) * SV_STR + row] = f2bf(fv.x);
        sVt[(d + 1) * SV_STR + row] = f2bf(fv.y);
        sVt[(d + 2) * SV_STR + row] = f2bf(fv.z);
        sVt[(d + 3) * SV_STR + row] = f2bf(fv.w);
    }
    __syncthreads();

    const int w    = tid >> 5;
    const int lane = tid & 31;
    const int half = lane >> 4;     // lane 0-15 vs 16-31
    const int l16  = lane & 15;

    // ---------------- Phase 1: S = Q * K^T  (64x256, raw) ----------------
    // 64 tiles of 16x16; tile t -> (mt = t/16, nt = t%16); each wave strides by 8.
    for (int t = w; t < 64; t += 8) {
        int mt = t >> 4, nt = t & 15;
        const unsigned short* qrow = sQ + (mt * 16 + l16) * SQ_STR;
        const unsigned short* krow = sK + (nt * 16 + l16) * SK_STR;
        v8f acc = {0.f,0.f,0.f,0.f,0.f,0.f,0.f,0.f};
#pragma unroll
        for (int kk = 0; kk < 4; ++kk) {
            int dbase = kk * 32;
            FragU A, Bf;
            // A (16x32 bf16): lane half h holds d-pairs {8h..8h+7} and {16+8h..16+8h+7}
            A.q[0]  = *(const uint4*)(qrow + dbase + 8 * half);
            A.q[1]  = *(const uint4*)(qrow + dbase + 16 + 8 * half);
            // B (32x16 bf16): lane half h holds K values {8h..8h+7} and {16+8h..16+8h+7}
            Bf.q[0] = *(const uint4*)(krow + dbase + 8 * half);
            Bf.q[1] = *(const uint4*)(krow + dbase + 16 + 8 * half);
            acc = __builtin_amdgcn_wmma_f32_16x16x32_bf16(false, A.v, false, Bf.v,
                                                          (short)0, acc, false, false);
        }
#pragma unroll
        for (int r = 0; r < 8; ++r)   // C layout: M = r + 8*half, N = l16
            sS[(mt * 16 + r + 8 * half) * SS_STR + nt * 16 + l16] = acc[r];
    }
    __syncthreads();

    // ---------------- Phase 2: masked softmax over 256 cols ----------------
    {
        const float scale = 0.08838834764831845f;   // 1/sqrt(128)
        int row = tid >> 2, sub = tid & 3;          // 4 lanes per query row
        int j   = row;
        int m   = g * WW + row;
        int ls  = m - (WW - 1); if (ls < 0) ls = 0; // local_start
        int c0  = sub * 64;

        // Each 64-col chunk is a single mask category; the kept set is a
        // contiguous [lo, hi] range in chunk-local coordinates.
        int lo, hi;
        if (sub == 0)      { lo = (g > 0) ? (j + 1) : 64; hi = 63; }   // local prev
        else if (sub == 1) { lo = 0; hi = j; }                          // local cur
        else if (sub == 2) { lo = 0; hi = (ls - 1) >> 6; }              // strided
        else               { lo = 0; hi = (ls - 64) >> 6; }             // rel

        float vals[64];
        float mx = -3.0e38f;
        const float4* srow = (const float4*)(sS + row * SS_STR + c0);
#pragma unroll
        for (int q4 = 0; q4 < 16; ++q4) {
            float4 f = srow[q4];
#pragma unroll
            for (int e = 0; e < 4; ++e) {
                int  cc = q4 * 4 + e;
                float s = ((const float*)&f)[e] * scale;
                s = (cc >= lo && cc <= hi) ? s : -1.0e30f;
                vals[cc] = s;
                mx = fmaxf(mx, s);
            }
        }
        mx = fmaxf(mx, __shfl_xor(mx, 1, 32));
        mx = fmaxf(mx, __shfl_xor(mx, 2, 32));
        float sum = 0.f;
#pragma unroll
        for (int cc = 0; cc < 64; ++cc) {
            float e = __expf(vals[cc] - mx);
            vals[cc] = e;
            sum += e;
        }
        sum += __shfl_xor(sum, 1, 32);
        sum += __shfl_xor(sum, 2, 32);
        float inv = 1.0f / sum;

        uint4* prow = (uint4*)(sP + row * SP_STR + c0);
#pragma unroll
        for (int q8 = 0; q8 < 8; ++q8) {
            uint4 pk;
            unsigned int* pu = (unsigned int*)&pk;
#pragma unroll
            for (int e = 0; e < 4; ++e) {
                int cc = q8 * 8 + e * 2;
                pu[e] = pack2_bf16(vals[cc] * inv, vals[cc + 1] * inv);
            }
            prow[q8] = pk;
        }
    }
    __syncthreads();

    // ---------------- Phase 3: O = P * V  (64x128) ----------------
    // 32 tiles of 16x16; tile t -> (mt = t/8, dt = t%8)
    for (int t = w; t < 32; t += 8) {
        int mt = t >> 3, dt = t & 7;
        const unsigned short* prow = sP  + (mt * 16 + l16) * SP_STR;
        const unsigned short* vrow = sVt + (dt * 16 + l16) * SV_STR;
        v8f acc = {0.f,0.f,0.f,0.f,0.f,0.f,0.f,0.f};
#pragma unroll
        for (int kk = 0; kk < 8; ++kk) {
            int kbase = kk * 32;
            FragU A, Bf;
            A.q[0]  = *(const uint4*)(prow + kbase + 8 * half);        // key pairs for P
            A.q[1]  = *(const uint4*)(prow + kbase + 16 + 8 * half);
            Bf.q[0] = *(const uint4*)(vrow + kbase + 8 * half);        // key pairs for Vt
            Bf.q[1] = *(const uint4*)(vrow + kbase + 16 + 8 * half);
            acc = __builtin_amdgcn_wmma_f32_16x16x32_bf16(false, A.v, false, Bf.v,
                                                          (short)0, acc, false, false);
        }
#pragma unroll
        for (int r = 0; r < 8; ++r)
            obh[(size_t)(g * WW + mt * 16 + r + 8 * half) * DD + dt * 16 + l16] = acc[r];
    }
}

extern "C" void kernel_launch(void* const* d_in, const int* in_sizes, int n_in,
                              void* d_out, int out_size, void* d_ws, size_t ws_size,
                              hipStream_t stream) {
    const float* q  = (const float*)d_in[0];
    const float* k  = (const float*)d_in[1];
    const float* v  = (const float*)d_in[2];
    const float* rk = (const float*)d_in[3];
    const float* rv = (const float*)d_in[4];
    float* out = (float*)d_out;

    dim3 grid(BB * HH * NBLK);   // 2048 workgroups: one per (b,h,query-block)
    dim3 block(256);             // 8 wave32
    cron_root_attn_wmma<<<grid, block, SMEM_BYTES, stream>>>(q, k, v, rk, rv, out);
}
